// BlockSparseRingMultiheadDilatedAttention_89687507076104
// MI455X (gfx1250) — matrix operand order repaired and checked by reading, not compiled, not run on hardware
//
#include <hip/hip_runtime.h>
#include <hip/hip_bf16.h>
#include <math.h>

typedef __attribute__((ext_vector_type(16))) _Float16     v16h;
typedef __attribute__((ext_vector_type(8)))  _Float16     v8h;
typedef __attribute__((ext_vector_type(8)))  float        v8f;
typedef __attribute__((ext_vector_type(4)))  unsigned int v4u;
typedef __attribute__((ext_vector_type(8)))  int          v8i_t;
typedef __attribute__((ext_vector_type(4)))  int          v4i_t;

#define EMBED 1024
#define NROWS 8192
#define HDIM  64
#define QBSEG 128   // 2048 dilated positions / 16 query rows

static __device__ __forceinline__ v8f wmma_f16(v16h a, v16h b, v8f c) {
  // D = A(16x32 f16) * B(32x16 f16) + C(16x16 f32)
  return __builtin_amdgcn_wmma_f32_16x16x32_f16(false, a, false, b, (short)0, c,
                                                false, false);
}

// ---------------------------------------------------------------------------
// Tensor Data Mover: DMA a 2-D tile (tile_dim1 rows x tile_dim0 f16 elements,
// row stride = row_stride_elems) from global memory into LDS at lds_byte_addr.
// Descriptor layout per CDNA5 ISA 8.3/8.4 (D# group0/group1), 2-D => groups
// 2/3 unused. Completion tracked with TENSORcnt.
// ---------------------------------------------------------------------------
static __device__ __forceinline__ void tdm_load_tile_f16(
    const _Float16* gsrc, unsigned lds_byte_addr,
    unsigned tile_d0, unsigned tile_d1, unsigned row_stride_elems)
{
  const unsigned long long ga = (unsigned long long)(size_t)gsrc;
  v4u g0;
  g0[0] = 1u;                                        // count=1, user descriptor
  g0[1] = lds_byte_addr;                             // LDS destination (bytes)
  g0[2] = (unsigned)(ga & 0xffffffffu);              // global_addr[31:0]
  g0[3] = (unsigned)((ga >> 32) & 0x01ffffffu)       // global_addr[56:32]
          | (2u << 30);                              // type=2 ("image")
  const unsigned dim0 = tile_d0;                     // tensor dims (OOB bound)
  const unsigned dim1 = tile_d1;
  const unsigned long long s0 = row_stride_elems;    // dim0 stride (elements)
  v8i_t g1;
  g1[0] = (int)(1u << 16);                           // data_size=1 -> 2 bytes
  g1[1] = (int)((dim0 & 0xffffu) << 16);             // abar_addr=0 | dim0[15:0]
  g1[2] = (int)((dim0 >> 16) | ((dim1 & 0xffffu) << 16));
  g1[3] = (int)((dim1 >> 16) | (tile_d0 << 16));     // dim1[31:16] | tile_dim0
  g1[4] = (int)(tile_d1 & 0xffffu);                  // tile_dim1 | tile_dim2=0
  g1[5] = (int)(s0 & 0xffffffffu);                   // dim0_stride[31:0]
  g1[6] = (int)((s0 >> 32) & 0xffffu);               // stride[47:32] | s1[15:0]
  g1[7] = 0;                                         // dim1_stride hi (unused)
  const v4i_t z4 = {0, 0, 0, 0};
#if __clang_major__ >= 23
  const v8i_t z8 = {0, 0, 0, 0, 0, 0, 0, 0};
  __builtin_amdgcn_tensor_load_to_lds(g0, g1, z4, z4, z8, 0);
#else
  __builtin_amdgcn_tensor_load_to_lds(g0, g1, z4, z4, 0);
#endif
}

// ---------------------------------------------------------------------------
// GEMM: Out[M,N] = A[M,K] * W^T + bias   (W stored [N,K] row-major)
// One wave computes a 32x32 tile. A-fragment: lane=row, halves at kb/kb+16.
// B-fragment: lane=column n, halves at kb/kb+16 (W row n contiguous in k).
// ---------------------------------------------------------------------------
template<bool A_IS_F16, bool OUT_IS_F16>
__global__ __launch_bounds__(256)
void gemm_wmma_kernel(const void* __restrict__ Aany,
                      const float* __restrict__ W,
                      const float* __restrict__ bias,
                      void* __restrict__ Outany,
                      int M, int N, int Kd, float scale)
{
  const int lane = threadIdx.x & 31;
  const int wid  = threadIdx.x >> 5;
  const int tiles_n = N >> 5;
  const int t  = blockIdx.x * 8 + wid;
  const int m0 = (t / tiles_n) << 5;
  const int n0 = (t % tiles_n) << 5;
  if (m0 >= M) return;

  const int ar  = lane & 15;
  const int akb = (lane < 16) ? 0 : 8;
  const int bn  = lane & 15;
  const int bko = (lane < 16) ? 0 : 16;

  v8f c00 = {}, c01 = {}, c10 = {}, c11 = {};

  for (int k0 = 0; k0 < Kd; k0 += 32) {
    v16h a0, a1, b0, b1;
    if (A_IS_F16) {
      const _Float16* A  = (const _Float16*)Aany;
      const _Float16* p0 = A + (size_t)(m0 + ar) * Kd + k0 + akb;
      const _Float16* p1 = A + (size_t)(m0 + 16 + ar) * Kd + k0 + akb;
      __builtin_prefetch(p0 + 32);                   // global_prefetch_b8
      v8h l0 = *(const v8h*)p0, h0 = *(const v8h*)(p0 + 16);
      v8h l1 = *(const v8h*)p1, h1 = *(const v8h*)(p1 + 16);
      a0 = __builtin_shufflevector(l0, h0, 0,1,2,3,4,5,6,7,8,9,10,11,12,13,14,15);
      a1 = __builtin_shufflevector(l1, h1, 0,1,2,3,4,5,6,7,8,9,10,11,12,13,14,15);
    } else {
      const float* A  = (const float*)Aany;
      const float* p0 = A + (size_t)(m0 + ar) * Kd + k0 + akb;
      const float* p1 = A + (size_t)(m0 + 16 + ar) * Kd + k0 + akb;
      __builtin_prefetch(p0 + 32);                   // global_prefetch_b8
#pragma unroll
      for (int e = 0; e < 8; ++e) {
        a0[e] = (_Float16)p0[e];  a0[8 + e] = (_Float16)p0[16 + e];
        a1[e] = (_Float16)p1[e];  a1[8 + e] = (_Float16)p1[16 + e];
      }
    }
    const float* w0 = W + (size_t)(n0 + bn) * Kd + k0 + bko;
    const float* w1 = W + (size_t)(n0 + 16 + bn) * Kd + k0 + bko;
    __builtin_prefetch(w0 + 32);                     // global_prefetch_b8
#pragma unroll
    for (int e = 0; e < 16; ++e) {
      b0[e] = (_Float16)w0[akb - bko + (e & 7) + ((e >> 3) << 4)];
      b1[e] = (_Float16)w1[akb - bko + (e & 7) + ((e >> 3) << 4)];
    }
    c00 = wmma_f16(a0, b0, c00);
    c01 = wmma_f16(a0, b1, c01);
    c10 = wmma_f16(a1, b0, c10);
    c11 = wmma_f16(a1, b1, c11);
  }

  const int rofs = (lane < 16) ? 0 : 8;
#pragma unroll
  for (int rt = 0; rt < 2; ++rt) {
#pragma unroll
    for (int bt = 0; bt < 2; ++bt) {
      v8f c = rt ? (bt ? c11 : c10) : (bt ? c01 : c00);
      const int col = n0 + bt * 16 + bn;
      const float bv = bias[col];
#pragma unroll
      for (int v = 0; v < 8; ++v) {
        const int row = m0 + rt * 16 + rofs + v;
        const float r = c[v] + bv;
        if (OUT_IS_F16)
          ((_Float16*)Outany)[(size_t)row * N + col] = (_Float16)(r * scale);
        else
          ((float*)Outany)[(size_t)row * N + col] = r;
      }
    }
  }
}

// ---------------------------------------------------------------------------
// Flash attention over dilated segments. One wave handles one
// (group, segment, head, 16-query-row block). m = 2048 dilated positions.
// V tiles are DMA'd into LDS by the Tensor Data Mover and fenced with
// s_wait_tensorcnt; P is re-laid-out C->A through per-wave LDS (in-order DS).
// ---------------------------------------------------------------------------
__global__ __launch_bounds__(128)
void attn_kernel(const _Float16* __restrict__ Q,
                 const _Float16* __restrict__ K,
                 const _Float16* __restrict__ V,
                 _Float16* __restrict__ AO)
{
  __shared__ __align__(32) _Float16 PldsAll[4][16 * 32];
  __shared__ __align__(32) _Float16 VldsAll[4][32 * 64];
  const int lane = threadIdx.x & 31;
  const int wid  = threadIdx.x >> 5;
  _Float16* Plds = PldsAll[wid];
  _Float16* Vlds = VldsAll[wid];
  const unsigned vlds_base = (unsigned)(size_t)&Vlds[0];   // LDS byte address

  const int unit = blockIdx.x * 4 + wid;
  const int G0 = 4 * 8 * QBSEG;                 // group0 units
  int g, s, h, qb;
  if (unit < G0) {
    g = 0; s = unit / (8 * QBSEG);
    int r = unit % (8 * QBSEG); h = r / QBSEG; qb = r % QBSEG;
  } else {
    g = 1; int u = unit - G0; s = u / (8 * QBSEG);
    int r = u % (8 * QBSEG); h = 8 + r / QBSEG; qb = r % QBSEG;
  }
  const int dil = g ? 2 : 1;
  const int off = g ? 1 : 0;                    // offset = g % r
  const int seg_base = s * (g ? 4096 : 2048);
  const int col0 = h * HDIM;
  const int q0 = qb * 16;

  const int ar   = lane & 15;
  const int akb  = (lane < 16) ? 0 : 8;
  const int bn   = lane & 15;
  const int bko  = (lane < 16) ? 0 : 16;
  const int rofs = (lane < 16) ? 0 : 8;

  // Q fragments (pre-scaled by 1/sqrt(64) at projection time): two D-chunks
  v16h aq0, aq1;
  {
    const size_t qoff = (size_t)(seg_base + off + dil * (q0 + ar)) * EMBED + col0;
    const _Float16* p = Q + qoff;
    aq0 = *(const v16h*)(p + bko);
    aq1 = *(const v16h*)(p + 32 + bko);
  }

  v8f o[4] = {v8f{}, v8f{}, v8f{}, v8f{}};
  float mrow[8], lrow[8];
#pragma unroll
  for (int v = 0; v < 8; ++v) { mrow[v] = -3.0e38f; lrow[v] = 0.0f; }

  const int jend = q0 + 16;                     // causal: keys 0..q0+15
  for (int j0 = 0; j0 < jend; j0 += 32) {
    // ---- TDM: DMA 32 (dilated) V rows x 64 halfs into this wave's LDS ----
    // Issued up front so the DMA overlaps S=QK^T and the softmax below.
    tdm_load_tile_f16(
        V + (size_t)(seg_base + off + dil * j0) * EMBED + col0,
        vlds_base, /*tile_d0=*/64, /*tile_d1=*/32,
        /*row_stride_elems=*/(unsigned)(dil * EMBED));

    // ---- S = Q * K^T for 32 key columns (two 16-col tiles) ----
    v16h bk00, bk01, bk10, bk11;
    {
      const size_t k0off = (size_t)(seg_base + off + dil * (j0 + bn)) * EMBED + col0;
      const size_t k1off = (size_t)(seg_base + off + dil * (j0 + 16 + bn)) * EMBED + col0;
      bk00 = *(const v16h*)(K + k0off + bko);
      bk01 = *(const v16h*)(K + k0off + 32 + bko);
      bk10 = *(const v16h*)(K + k1off + bko);
      bk11 = *(const v16h*)(K + k1off + 32 + bko);
    }
    v8f z = {};
    v8f s0 = wmma_f16(aq0, bk00, z); s0 = wmma_f16(aq1, bk01, s0);
    v8f s1 = wmma_f16(aq0, bk10, z); s1 = wmma_f16(aq1, bk11, s1);

    // ---- online softmax (row = q0 + rofs + v; cols j0+bn, j0+16+bn) ----
    float rowscale[8];
#pragma unroll
    for (int v = 0; v < 8; ++v) {
      const int qi  = q0 + rofs + v;
      const int ki0 = j0 + bn;
      const int ki1 = ki0 + 16;
      float e0 = (ki0 <= qi) ? s0[v] : -3.0e38f;
      float e1 = (ki1 <= qi) ? s1[v] : -3.0e38f;
      float mx = fmaxf(e0, e1);
#pragma unroll
      for (int d = 1; d < 16; d <<= 1) mx = fmaxf(mx, __shfl_xor(mx, d, 32));
      const float mn = fmaxf(mrow[v], mx);
      const float sc = __expf(mrow[v] - mn);
      const float p0 = __expf(e0 - mn);
      const float p1 = __expf(e1 - mn);
      float rs = p0 + p1;
#pragma unroll
      for (int d = 1; d < 16; d <<= 1) rs += __shfl_xor(rs, d, 32);
      lrow[v] = lrow[v] * sc + rs;
      mrow[v] = mn;
      rowscale[v] = sc;
      // C-layout -> row-major LDS tile (16x32) for the PV A-fragment
      Plds[(rofs + v) * 32 + bn]      = (_Float16)p0;
      Plds[(rofs + v) * 32 + 16 + bn] = (_Float16)p1;
    }

    // ---- P as A-fragment (per-wave LDS ops are in-order; no barrier) ----
    v16h pf;
    {
      v8h lo = *(const v8h*)&Plds[ar * 32 + akb];
      v8h hi = *(const v8h*)&Plds[ar * 32 + akb + 16];
      pf = __builtin_shufflevector(lo, hi, 0,1,2,3,4,5,6,7,8,9,10,11,12,13,14,15);
    }

    // ---- fence the TDM tile, then O = O*scale + P*V (four 16-col tiles) ----
    __builtin_amdgcn_s_wait_tensorcnt(0);
#pragma unroll
    for (int t = 0; t < 4; ++t) {
      v16h bv;
      const int dcol = t * 16 + bn;
#pragma unroll
      for (int e = 0; e < 16; ++e) bv[e] = Vlds[(akb + (e & 7) + ((e >> 3) << 4)) * 64 + dcol];
      v8f oo = o[t];
#pragma unroll
      for (int v = 0; v < 8; ++v) oo[v] *= rowscale[v];
      o[t] = wmma_f16(pf, bv, oo);
    }
  }

  // ---- finalize: O / l, scatter to dilated rows of AO ----
  float inv[8];
#pragma unroll
  for (int v = 0; v < 8; ++v) inv[v] = (lrow[v] > 0.0f) ? 1.0f / lrow[v] : 0.0f;
#pragma unroll
  for (int t = 0; t < 4; ++t) {
#pragma unroll
    for (int v = 0; v < 8; ++v) {
      const size_t row = (size_t)(seg_base + off + dil * (q0 + rofs + v));
      AO[row * EMBED + col0 + t * 16 + bn] = (_Float16)(o[t][v] * inv[v]);
    }
  }
}

__global__ __launch_bounds__(256)
void zero_kernel(unsigned int* __restrict__ p, size_t n)
{
  size_t i = (size_t)blockIdx.x * blockDim.x + threadIdx.x;
  const size_t stride = (size_t)gridDim.x * blockDim.x;
  for (; i < n; i += stride) p[i] = 0u;
}

extern "C" void kernel_launch(void* const* d_in, const int* in_sizes, int n_in,
                              void* d_out, int out_size, void* d_ws, size_t ws_size,
                              hipStream_t stream)
{
  (void)in_sizes; (void)n_in; (void)out_size; (void)ws_size;
  const float* Xq = (const float*)d_in[0];
  const float* Xk = (const float*)d_in[1];
  const float* Xv = (const float*)d_in[2];
  const float* Wq = (const float*)d_in[3];
  const float* bq = (const float*)d_in[4];
  const float* Wk = (const float*)d_in[5];
  const float* bk = (const float*)d_in[6];
  const float* Wv = (const float*)d_in[7];
  const float* bv = (const float*)d_in[8];
  const float* Wo = (const float*)d_in[9];
  const float* bo = (const float*)d_in[10];
  float* out = (float*)d_out;

  _Float16* Q16 = (_Float16*)d_ws;
  _Float16* K16 = Q16 + (size_t)NROWS * EMBED;
  _Float16* V16 = K16 + (size_t)NROWS * EMBED;
  _Float16* AO  = V16 + (size_t)NROWS * EMBED;

  const int gemm_blocks = (NROWS / 32) * (EMBED / 32) / 8;   // 1024

  // Projections (fp32 in-register -> f16, f32 WMMA accumulate, bias epilogue).
  // Score scale 1/sqrt(64) folded into Q.
  gemm_wmma_kernel<false, true><<<gemm_blocks, 256, 0, stream>>>(
      Xq, Wq, bq, Q16, NROWS, EMBED, EMBED, 0.125f);
  gemm_wmma_kernel<false, true><<<gemm_blocks, 256, 0, stream>>>(
      Xk, Wk, bk, K16, NROWS, EMBED, EMBED, 1.0f);
  gemm_wmma_kernel<false, true><<<gemm_blocks, 256, 0, stream>>>(
      Xv, Wv, bv, V16, NROWS, EMBED, EMBED, 1.0f);

  // Dilation-2 heads only write half the rows -> zero the attention buffer.
  zero_kernel<<<1024, 256, 0, stream>>>((unsigned int*)AO,
                                        (size_t)NROWS * EMBED / 2);

  // group0: 4 segs * 8 heads * 128 qblocks, group1: 2 * 8 * 128 => 6144 waves
  const int units = 4 * 8 * QBSEG + 2 * 8 * QBSEG;
  attn_kernel<<<units / 4, 128, 0, stream>>>(Q16, K16, V16, AO);

  // Output projection back to fp32 with bias.
  gemm_wmma_kernel<true, false><<<gemm_blocks, 256, 0, stream>>>(
      AO, Wo, bo, out, NROWS, EMBED, EMBED, 1.0f);
}
